// RationaleSelectorModel_64647847739951
// MI455X (gfx1250) — matrix operand order repaired and checked by reading, not compiled, not run on hardware
//
#include <hip/hip_runtime.h>
#include <hip/hip_bf16.h>

// ---- problem constants (match reference setup_inputs) ----
constexpr int B_ = 8, L_ = 2048, D_ = 768, H_ = 1024, R_ = 3;
constexpr int M_ = B_ * L_;          // 16384 tokens
#define LN_EPS_F   1e-5f
#define TAU_RANK_F 0.5f

typedef _Float16 half8 __attribute__((ext_vector_type(8)));
typedef _Float16 v16h  __attribute__((ext_vector_type(16)));
typedef float    v8f   __attribute__((ext_vector_type(8)));
typedef int      v4i_  __attribute__((vector_size(16)));

#if defined(__has_builtin)
#if __has_builtin(__builtin_amdgcn_global_load_async_to_lds_b128) && \
    __has_builtin(__builtin_amdgcn_s_wait_asynccnt)
#define HAVE_ASYNC_LDS 1
#endif
#endif

#ifdef HAVE_ASYNC_LDS
// param1: v4i* in addrspace(1) (global), param2: v4i* in addrspace(3) (LDS)
#define GAS1(p) ((__attribute__((address_space(1))) v4i_*)(p))
#define LAS3(p) ((__attribute__((address_space(3))) v4i_*)(p))
#endif

// ---------------- block reduce helper (256 threads) ----------------
__device__ __forceinline__ float blockReduce256(float v, float* red) {
    int t = threadIdx.x;
    red[t] = v; __syncthreads();
    for (int o = 128; o > 0; o >>= 1) {
        if (t < o) red[t] += red[t + o];
        __syncthreads();
    }
    float r = red[0];
    __syncthreads();
    return r;
}

// ---------------- K1: mask + LayerNorm + f16 convert ----------------
__global__ __launch_bounds__(256) void ln_mask_kernel(
    const float* __restrict__ emb, const float* __restrict__ attn,
    const float* __restrict__ lng, const float* __restrict__ lnb,
    _Float16* __restrict__ xo)
{
    __shared__ float red[256];
    int m = blockIdx.x, t = threadIdx.x;
    float a = attn[m];
    const float* e = emb + (size_t)m * D_;
    float x0 = e[t] * a, x1 = e[t + 256] * a, x2 = e[t + 512] * a;
    float mu = blockReduce256(x0 + x1 + x2, red) * (1.0f / (float)D_);
    float d0 = x0 - mu, d1 = x1 - mu, d2 = x2 - mu;
    float var = blockReduce256(d0 * d0 + d1 * d1 + d2 * d2, red) * (1.0f / (float)D_);
    float inv = rsqrtf(var + LN_EPS_F);
    _Float16* xm = xo + (size_t)m * D_;
    xm[t]       = (_Float16)(d0 * inv * lng[t]       + lnb[t]);
    xm[t + 256] = (_Float16)(d1 * inv * lng[t + 256] + lnb[t + 256]);
    xm[t + 512] = (_Float16)(d2 * inv * lng[t + 512] + lnb[t + 512]);
}

// ---------------- K1b: w1 [D,H] -> w1t f16 [H,D] ----------------
__global__ void w1t_kernel(const float* __restrict__ w1, _Float16* __restrict__ w1t) {
    int idx = blockIdx.x * 256 + threadIdx.x;
    if (idx < D_ * H_) {
        int k = idx / H_, n = idx % H_;
        w1t[(size_t)n * D_ + k] = (_Float16)w1[idx];
    }
}

// ---------------- K2: WMMA GEMM + fused bias/GELU/w2-dot ----------------
// C[M,H] = Xf16[M,D] * W1t[H,D]^T ; epilogue: partial[m, ntile] = sum_n gelu(c+b1)*w2
// Double-buffered LDS tiles; async global->LDS staging when the toolchain has it.
#define KT 32
#define LPAD 8
#define NSTEP (D_ / KT)
__global__ __launch_bounds__(256) void gemm_gelu_kernel(
    const _Float16* __restrict__ xA, const _Float16* __restrict__ w1t,
    const float* __restrict__ b1, const float* __restrict__ w2,
    float* __restrict__ partial)
{
    __shared__ _Float16 At[2][128][KT + LPAD];
    __shared__ _Float16 Bt[2][128][KT + LPAD];
    __shared__ float s_part[128];

    int t = threadIdx.x;
    int wave = t >> 5, lane = t & 31;
    int lane15 = lane & 15;
    int hi = (lane >= 16);
    int tileM = blockIdx.x * 128;
    int tileN = blockIdx.y * 128;
    int lrow = t >> 1, lseg = t & 1;

    if (t < 128) s_part[t] = 0.0f;

    const _Float16* gA = xA  + (size_t)(tileM + lrow) * D_ + lseg * 16;
    const _Float16* gB = w1t + (size_t)(tileN + lrow) * D_ + lseg * 16;

    // stage one 128x32 A tile + 128x32 B tile into LDS buffer `buf` for K-chunk k0
    auto stage = [&](int buf, int k0) {
        _Float16* la = &At[buf][lrow][lseg * 16];
        _Float16* lb = &Bt[buf][lrow][lseg * 16];
#ifdef HAVE_ASYNC_LDS
        __builtin_amdgcn_global_load_async_to_lds_b128(GAS1(gA + k0),     LAS3(la),     0, 0);
        __builtin_amdgcn_global_load_async_to_lds_b128(GAS1(gA + k0 + 8), LAS3(la + 8), 0, 0);
        __builtin_amdgcn_global_load_async_to_lds_b128(GAS1(gB + k0),     LAS3(lb),     0, 0);
        __builtin_amdgcn_global_load_async_to_lds_b128(GAS1(gB + k0 + 8), LAS3(lb + 8), 0, 0);
#else
        const half8* sa = (const half8*)(gA + k0);
        half8 a0 = sa[0], a1 = sa[1];
        *(half8*)la       = a0;
        *(half8*)(la + 8) = a1;
        const half8* sb = (const half8*)(gB + k0);
        half8 b0 = sb[0], b1x = sb[1];
        *(half8*)lb       = b0;
        *(half8*)(lb + 8) = b1x;
#endif
    };

    v8f acc[8];
#pragma unroll
    for (int j = 0; j < 8; ++j) acc[j] = (v8f){0.f,0.f,0.f,0.f,0.f,0.f,0.f,0.f};

    stage(0, 0);
#ifdef HAVE_ASYNC_LDS
    __builtin_amdgcn_s_wait_asynccnt(0);
#endif
    __syncthreads();

    int arow = wave * 16 + lane15;
    int klo  = hi ? 8 : 0;
    int kblo = hi ? 8 : 0;

    for (int step = 0; step < NSTEP; ++step) {
        int cur = step & 1;
        if (step + 1 < NSTEP) stage(1 - cur, (step + 1) * KT);

        // A fragment: 16x32 f16 per ISA layout (lanes<16: K0-7 & K16-23; lanes>=16: K8-15 & K24-31)
        half8 fa0 = *(const half8*)&At[cur][arow][klo];
        half8 fa1 = *(const half8*)&At[cur][arow][16 + klo];
        v16h afrag = __builtin_shufflevector(fa0, fa1, 0,1,2,3,4,5,6,7,8,9,10,11,12,13,14,15);

#pragma unroll
        for (int j = 0; j < 8; ++j) {
            int bn = j * 16 + lane15;
            half8 fb0 = *(const half8*)&Bt[cur][bn][kblo];
            half8 fb1 = *(const half8*)&Bt[cur][bn][kblo + 16];
            v16h bfrag = __builtin_shufflevector(fb0, fb1, 0,1,2,3,4,5,6,7,8,9,10,11,12,13,14,15);
            acc[j] = __builtin_amdgcn_wmma_f32_16x16x32_f16(
                false, afrag, false, bfrag, (short)0, acc[j], false, false);
        }
#ifdef HAVE_ASYNC_LDS
        __builtin_amdgcn_s_wait_asynccnt(0);
#endif
        __syncthreads();
    }

    // epilogue: gelu(c + b1) * w2, reduce over this block's 128 columns
    float bv[8], wv[8];
#pragma unroll
    for (int j = 0; j < 8; ++j) {
        int n = tileN + j * 16 + lane15;
        bv[j] = b1[n]; wv[j] = w2[n];
    }
#pragma unroll
    for (int v = 0; v < 8; ++v) {
        float contrib = 0.0f;
#pragma unroll
        for (int j = 0; j < 8; ++j) {
            float h = acc[j][v] + bv[j];
            float gel = 0.5f * h * (1.0f + erff(h * 0.70710678118654752f));
            contrib += gel * wv[j];
        }
        int row = wave * 16 + v + (hi ? 8 : 0);
        atomicAdd(&s_part[row], contrib);
    }
    __syncthreads();
    if (t < 128)
        partial[(size_t)(tileM + t) * (H_ / 128) + blockIdx.y] = s_part[t];
}

// ---------------- K3: finalize scores ----------------
__global__ void scores_kernel(const float* __restrict__ partial,
                              const float* __restrict__ attn,
                              const float* __restrict__ b2,
                              float* __restrict__ scores)
{
    int m = blockIdx.x * 256 + threadIdx.x;
    if (m >= M_) return;
    float s = b2[0];
#pragma unroll
    for (int j = 0; j < H_ / 128; ++j) s += partial[(size_t)m * (H_ / 128) + j];
    scores[m] = (attn[m] != 0.0f) ? s : 0.0f;
}

// ---------------- K4: soft-rank + rank-pos + gates (one block per batch) ----------------
__global__ __launch_bounds__(256) void rank_gate_kernel(
    const float* __restrict__ scores, const float* __restrict__ attn,
    const float* __restrict__ rhos,
    float* __restrict__ zout, float* __restrict__ gout,
    float* __restrict__ gsum, float* __restrict__ denomB)
{
    __shared__ float sc[L_];
    __shared__ float at[L_];
    __shared__ float rk[L_];
    __shared__ float red[256];
    int b = blockIdx.x, t = threadIdx.x;

    float asum_l = 0.f, ssum_l = 0.f;
    for (int i = t; i < L_; i += 256) {
        float a = attn[b * L_ + i], s = scores[b * L_ + i];
        at[i] = a; sc[i] = s;
        asum_l += a; ssum_l += a * s;
    }
    __syncthreads();
    float asum = blockReduce256(asum_l, red);
    float ssum = blockReduce256(ssum_l, red);
    float denom = fmaxf(asum, 1.0f);
    float mean = ssum / denom;

    float vloc = 0.f;
    for (int i = t; i < L_; i += 256) { float d = sc[i] - mean; vloc += d * d * at[i]; }
    float var = blockReduce256(vloc, red) / denom;
    float sinv = rsqrtf(var + 1e-6f);
    for (int i = t; i < L_; i += 256) sc[i] = (sc[i] - mean) * sinv;
    __syncthreads();

    // soft ranks: r_i = 0.5 + attn_i * sum_j attn_j * sigmoid(-(s_i - s_j)/tau)
    for (int i = t; i < L_; i += 256) {
        float si = sc[i];
        float sum = 0.f;
        for (int j = 0; j < L_; ++j)
            sum += at[j] / (1.0f + __expf((si - sc[j]) * (1.0f / TAU_RANK_F)));
        float r = 0.5f + at[i] * sum;
        rk[i] = (at[i] == 0.0f) ? 1e9f : r;
    }
    __syncthreads();

    // rank position = stable-argsort counting rank
    int rp[L_ / 256];
    for (int c = 0; c < L_ / 256; ++c) {
        int i = t + c * 256;
        float ri = rk[i];
        int cnt = 0;
        for (int j = 0; j < L_; ++j) {
            float rj = rk[j];
            cnt += (rj < ri) || ((rj == ri) && (j < i));
        }
        rp[c] = cnt;
    }
    if (t == 0) denomB[b] = denom;

    float Le = asum;
    for (int r = 0; r < R_; ++r) {
        float kr = rintf(rhos[r] * Le);       // jnp.round = RNE
        int ki = (Le > 0.0f) ? ((int)kr < 1 ? 1 : (int)kr) : 0;
        float kf = (float)ki;

        float grw[L_ / 256];
        float loc = 0.f;
        for (int c = 0; c < L_ / 256; ++c) {
            int i = t + c * 256;
            float x = kf - rk[i];
            float graw = at[i] / (1.0f + __expf(-x));
            grw[c] = graw; loc += graw;
        }
        float tot = blockReduce256(loc, red);
        float scale = kf / fmaxf(tot, 1e-8f);

        float gl = 0.f;
        for (int c = 0; c < L_ / 256; ++c) {
            int i = t + c * 256;
            float z = grw[c] * scale;
            zout[((size_t)r * B_ + b) * L_ + i] = z;
            float g = (rp[c] < ki) ? 1.0f : 0.0f;
            gout[((size_t)r * B_ + b) * L_ + i] = g;
            gl += at[i] * g;
        }
        float gs = blockReduce256(gl, red);
        if (t == 0) gsum[r * B_ + b] = gs;
    }
}

// ---------------- K5: full_rep masked-mean pool ----------------
__global__ void fullrep_kernel(const float* __restrict__ emb, const float* __restrict__ attn,
                               const float* __restrict__ denomB, float* __restrict__ full)
{
    int d = blockIdx.x * 256 + threadIdx.x;   // grid.x = D/256
    int b = blockIdx.y;
    float acc = 0.f;
    for (int l = 0; l < L_; ++l)
        acc += emb[((size_t)b * L_ + l) * D_ + d] * attn[b * L_ + l];
    full[(size_t)b * D_ + d] = acc / denomB[b];
}

// ---------------- K6: pred_rep pooled gather (all R in one pass) ----------------
__global__ void predrep_kernel(const int* __restrict__ ids, const float* __restrict__ tab,
                               const float* __restrict__ attn, const float* __restrict__ gout,
                               const float* __restrict__ gsum, float* __restrict__ pred)
{
    int d = blockIdx.x * 256 + threadIdx.x;
    int b = blockIdx.y;
    float a0 = 0.f, a1 = 0.f, a2 = 0.f;
    for (int l = 0; l < L_; ++l) {
        int id = ids[b * L_ + l];
        if (l + 1 < L_)
            __builtin_prefetch(&tab[(size_t)ids[b * L_ + l + 1] * D_ + d], 0, 1);
        float tv = tab[(size_t)id * D_ + d];
        float aa = attn[b * L_ + l];
        float e0 = aa * gout[((size_t)0 * B_ + b) * L_ + l];
        float e1 = aa * gout[((size_t)1 * B_ + b) * L_ + l];
        float e2 = aa * gout[((size_t)2 * B_ + b) * L_ + l];
        a0 += tv * e0; a1 += tv * e1; a2 += tv * e2;
    }
    pred[((size_t)0 * B_ + b) * D_ + d] = a0 / fmaxf(gsum[0 * B_ + b], 1.0f);
    pred[((size_t)1 * B_ + b) * D_ + d] = a1 / fmaxf(gsum[1 * B_ + b], 1.0f);
    pred[((size_t)2 * B_ + b) * D_ + d] = a2 / fmaxf(gsum[2 * B_ + b], 1.0f);
}

// ---------------- K7: cosine per (r,b), then mean loss ----------------
__global__ void cos_kernel(const float* __restrict__ pred, const float* __restrict__ full,
                           float* __restrict__ cosbuf)
{
    __shared__ float red[256];
    int rb = blockIdx.x;         // [0, R*B)
    int b = rb % B_;
    int t = threadIdx.x;
    float num = 0.f, np = 0.f, nf = 0.f;
    for (int d = t; d < D_; d += 256) {
        float p = pred[(size_t)rb * D_ + d];
        float f = full[(size_t)b * D_ + d];
        num += p * f; np += p * p; nf += f * f;
    }
    float Num = blockReduce256(num, red);
    float Np  = blockReduce256(np,  red);
    float Nf  = blockReduce256(nf,  red);
    if (t == 0) {
        float den = fmaxf(sqrtf(Np), 1e-8f) * fmaxf(sqrtf(Nf), 1e-8f);
        cosbuf[rb] = Num / den;
    }
}

__global__ void loss_kernel(const float* __restrict__ cosbuf, float* __restrict__ out_loss)
{
    if (threadIdx.x == 0) {
        float s = 0.f;
        for (int i = 0; i < R_ * B_; ++i) s += 1.0f - cosbuf[i];
        out_loss[0] = s / (float)(R_ * B_);
    }
}

// ---------------- host launcher ----------------
extern "C" void kernel_launch(void* const* d_in, const int* in_sizes, int n_in,
                              void* d_out, int out_size, void* d_ws, size_t ws_size,
                              hipStream_t stream)
{
    const int*   ids  = (const int*)  d_in[0];
    const float* emb  = (const float*)d_in[1];
    const float* attn = (const float*)d_in[2];
    const float* rhos = (const float*)d_in[3];
    const float* lng  = (const float*)d_in[4];
    const float* lnb  = (const float*)d_in[5];
    const float* w1   = (const float*)d_in[6];
    const float* b1   = (const float*)d_in[7];
    const float* w2   = (const float*)d_in[8];
    const float* b2   = (const float*)d_in[9];
    const float* tab  = (const float*)d_in[10];

    float* out = (float*)d_out;
    float* zout = out;
    float* gout = out + (size_t)R_ * B_ * L_;
    float* lossp = out + 2 * (size_t)R_ * B_ * L_;

    // workspace carve-up (256B aligned)
    char* w = (char*)d_ws;
    size_t off = 0;
    auto carve = [&](size_t bytes) {
        char* p = w + off;
        off = (off + bytes + 255) & ~(size_t)255;
        return p;
    };
    _Float16* xf16   = (_Float16*)carve((size_t)M_ * D_ * 2);     // 24 MB
    _Float16* w1t    = (_Float16*)carve((size_t)H_ * D_ * 2);     // 1.5 MB
    float*    part   = (float*)   carve((size_t)M_ * (H_/128) * 4);
    float*    scores = (float*)   carve((size_t)M_ * 4);
    float*    denomB = (float*)   carve((size_t)B_ * 4);
    float*    gsum   = (float*)   carve((size_t)R_ * B_ * 4);
    float*    full   = (float*)   carve((size_t)B_ * D_ * 4);
    float*    pred   = (float*)   carve((size_t)R_ * B_ * D_ * 4);
    float*    cosbuf = (float*)   carve((size_t)R_ * B_ * 4);
    (void)ws_size; (void)in_sizes; (void)n_in; (void)out_size;

    ln_mask_kernel<<<M_, 256, 0, stream>>>(emb, attn, lng, lnb, xf16);
    w1t_kernel<<<(D_ * H_ + 255) / 256, 256, 0, stream>>>(w1, w1t);
    gemm_gelu_kernel<<<dim3(M_ / 128, H_ / 128), 256, 0, stream>>>(xf16, w1t, b1, w2, part);
    scores_kernel<<<M_ / 256, 256, 0, stream>>>(part, attn, b2, scores);
    rank_gate_kernel<<<B_, 256, 0, stream>>>(scores, attn, rhos, zout, gout, gsum, denomB);
    fullrep_kernel<<<dim3(D_ / 256, B_), 256, 0, stream>>>(emb, attn, denomB, full);
    predrep_kernel<<<dim3(D_ / 256, B_), 256, 0, stream>>>(ids, tab, attn, gout, gsum, pred);
    cos_kernel<<<R_ * B_, 256, 0, stream>>>(pred, full, cosbuf);
    loss_kernel<<<1, 32, 0, stream>>>(cosbuf, lossp);
}
